// SemanticGrouping_67946382623492
// MI455X (gfx1250) — compile-verified, matrix operand order checked by reading, not run on hardware
//
#include <hip/hip_runtime.h>

typedef _Float16 v16h __attribute__((ext_vector_type(16)));
typedef float    v8f  __attribute__((ext_vector_type(8)));

#define B_  64
#define L_  1024
#define K_  2048
#define D_  768
#define INV_T 14.285714285714286f   // 1/0.07
#define EPS_ 1e-12f
#define PAD 36                       // padded LDS row stride (halves): 72B, 8B-aligned, conflict-friendly

__device__ __forceinline__ float waveSum(float x) {
#pragma unroll
  for (int off = 16; off > 0; off >>= 1) x += __shfl_xor(x, off, 32);
  return x;
}

// ---------------- normalization: wave per row ----------------
__global__ void proto_norm_kernel(const float* __restrict__ P, _Float16* __restrict__ Pn) {
  const int lane = threadIdx.x & 31;
  const int row  = blockIdx.x * 8 + (threadIdx.x >> 5);      // 0..K_-1
  const float* src = P + (size_t)row * D_;
  float s = 0.f;
  for (int c = lane; c < D_; c += 32) { float v = src[c]; s += v * v; }
  s = waveSum(s);
  const float scale = 1.0f / fmaxf(sqrtf(s), EPS_);
  _Float16* dst = Pn + (size_t)row * D_;
  for (int c = lane; c < D_; c += 32) dst[c] = (_Float16)(src[c] * scale);
}

__global__ void key_norm_kernel(const float* __restrict__ X, _Float16* __restrict__ Xn,
                                _Float16* __restrict__ Kh) {
  const int lane = threadIdx.x & 31;
  const int row  = blockIdx.x * 8 + (threadIdx.x >> 5);      // 0..B_*L_-1
  const float* src = X + (size_t)row * D_;
  float s = 0.f;
  for (int c = lane; c < D_; c += 32) { float v = src[c]; s += v * v; }
  s = waveSum(s);
  const float scale = 1.0f / fmaxf(sqrtf(s), EPS_);
  _Float16* dn = Xn + (size_t)row * D_;
  _Float16* dr = Kh + (size_t)row * D_;
  for (int c = lane; c < D_; c += 32) {
    float v = src[c];
    dn[c] = (_Float16)(v * scale);
    dr[c] = (_Float16)v;
  }
}

// ---------------- GEMM1: dots[b,k,l] = Pn[k,:] . Xn[b,l,:] ----------------
// block tile: 64 (k) x 128 (l); 8 waves, each 32x32; loop D in steps of 32.
// Staging is straight-line: all global loads issued, then all LDS stores.
__global__ void dots_gemm_kernel(const _Float16* __restrict__ Pn,
                                 const _Float16* __restrict__ Xn,
                                 float* __restrict__ dots) {
  __shared__ __align__(16) _Float16 sA[64 * PAD];    // Pn tile [k][d]
  __shared__ __align__(16) _Float16 sB[128 * PAD];   // Xn tile [l][d]
  const int b  = blockIdx.z;
  const int k0 = blockIdx.y * 64;
  const int l0 = blockIdx.x * 128;
  const int t = threadIdx.x;
  __builtin_assume(t < 256);
  const int lane = t & 31, wave = t >> 5;
  const int kq = wave >> 2, lq = wave & 3;           // wave quadrant
  const int m = lane & 15, g = lane >> 4;

  // staging map: chunk = 4 halves; row = t>>3 (+32*i), col chunk = (t&7)*4
  const int srow = t >> 3, spart = (t & 7) * 4;
  const _Float16* gA = Pn + (size_t)(k0 + srow) * D_ + spart;            // +32 rows for 2nd chunk
  const _Float16* gB = Xn + ((size_t)b * L_ + l0 + srow) * D_ + spart;   // +32*i rows
  _Float16* lA = sA + srow * PAD + spart;
  _Float16* lB = sB + srow * PAD + spart;

  v8f acc[2][2] = {};
  for (int d0 = 0; d0 < D_; d0 += 32) {
    // issue all 6 global loads first
    uint2 a0 = *(const uint2*)(gA + d0);
    uint2 a1 = *(const uint2*)(gA + (size_t)32 * D_ + d0);
    uint2 b0 = *(const uint2*)(gB + d0);
    uint2 b1 = *(const uint2*)(gB + (size_t)32 * D_ + d0);
    uint2 b2 = *(const uint2*)(gB + (size_t)64 * D_ + d0);
    uint2 b3 = *(const uint2*)(gB + (size_t)96 * D_ + d0);
    // then drain into LDS
    *(uint2*)(lA)            = a0;
    *(uint2*)(lA + 32 * PAD) = a1;
    *(uint2*)(lB)            = b0;
    *(uint2*)(lB + 32 * PAD) = b1;
    *(uint2*)(lB + 64 * PAD) = b2;
    *(uint2*)(lB + 96 * PAD) = b3;
    __syncthreads();

    v16h af[2], bf[2];
#pragma unroll
    for (int i = 0; i < 2; ++i) {
      unsigned* au = (unsigned*)&af[i];
      const _Float16* base = sA + (kq * 32 + i * 16 + m) * PAD;
#pragma unroll
      for (int r = 0; r < 8; ++r) {
        const int kk = (r < 4) ? (g * 8 + r * 2) : (16 + g * 8 + (r - 4) * 2);
        au[r] = *(const unsigned*)(base + kk);
      }
    }
#pragma unroll
    for (int j = 0; j < 2; ++j) {
      unsigned* bu = (unsigned*)&bf[j];
      const _Float16* base = sB + (lq * 32 + j * 16 + m) * PAD;
#pragma unroll
      for (int r = 0; r < 8; ++r) {
        const int kk = (r < 4) ? (g * 8 + r * 2) : (16 + g * 8 + (r - 4) * 2);
        bu[r] = *(const unsigned*)(base + kk);
      }
    }
#pragma unroll
    for (int i = 0; i < 2; ++i)
#pragma unroll
      for (int j = 0; j < 2; ++j)
        acc[i][j] = __builtin_amdgcn_wmma_f32_16x16x32_f16(
            false, af[i], false, bf[j], (short)0, acc[i][j], false, false);
    __syncthreads();
  }
  // epilogue: C layout — lane: col = lane&15 (+16 via g), VGPR r: row = g*8+r
#pragma unroll
  for (int i = 0; i < 2; ++i)
#pragma unroll
    for (int j = 0; j < 2; ++j) {
      const int l = l0 + lq * 32 + j * 16 + m;
#pragma unroll
      for (int r = 0; r < 8; ++r) {
        const int k = k0 + kq * 32 + i * 16 + g * 8 + r;
        dots[((size_t)b * K_ + k) * L_ + l] = acc[i][j][r];
      }
    }
}

// ---------------- online softmax stats over K axis, per (b,l) ----------------
__global__ void col_softmax_kernel(const float* __restrict__ dots,
                                   float* __restrict__ colm, float* __restrict__ colz) {
  const int idx = blockIdx.x * 256 + threadIdx.x;     // B_*L_ threads
  const int b = idx >> 10, l = idx & (L_ - 1);
  const float* col = dots + (size_t)b * K_ * L_ + l;
  float mmax = -3.0e38f, z = 0.f;
  for (int k = 0; k < K_; ++k) {
    const float v = col[(size_t)k * L_] * INV_T;
    if (v > mmax) { z = z * __expf(mmax - v) + 1.0f; mmax = v; }
    else          { z += __expf(v - mmax); }
  }
  colm[idx] = mmax;
  colz[idx] = z;
}

// ---------------- GEMM2: slots[b,k,d] = (sum_l attn_un[k,l]*key[l,d]) / (sum_l attn_un[k,l])
// block tile: 64 (k) x 128 (d); loop L in steps of 32; attn tile rebuilt in LDS as f16;
// per-row denominators accumulated in LDS with ds_add_f32, applied in epilogue.
__global__ void slots_gemm_kernel(const float* __restrict__ dots,
                                  const _Float16* __restrict__ Kh,
                                  const float* __restrict__ colm, const float* __restrict__ colz,
                                  float* __restrict__ slots) {
  __shared__ __align__(16) _Float16 sAttn[64 * PAD];    // [k][l] 64x32
  __shared__ __align__(16) _Float16 sKeyT[128 * PAD];   // [d][l] 128x32 (transposed stage)
  __shared__ float sm[32], srz[32], sDen[64];
  const int b  = blockIdx.z;
  const int k0 = blockIdx.y * 64;
  const int d0 = blockIdx.x * 128;
  const int t = threadIdx.x;
  __builtin_assume(t < 256);
  const int lane = t & 31, wave = t >> 5;
  const int kq = wave & 3, dq = wave >> 2;              // 4 k-strips x 2 d-halves(64 each)
  const int m = lane & 15, g = lane >> 4;

  if (t < 64) sDen[t] = 0.f;

  // attn staging map: thread t owns row=t>>2, cols [(t&3)*8, +8)
  const int arow = t >> 2, ac0 = (t & 3) * 8;
  const float* dRow = dots + ((size_t)b * K_ + k0 + arow) * L_ + ac0;

  // key staging map: 1024 uint2 chunks (32 l-rows x 32 d-chunks of 4); thread does 4 chunks
  const int klr = t >> 5;                 // base l-row 0..7 (+8*i)
  const int kdc = (t & 31) * 4;           // d chunk start
  const _Float16* gK = Kh + ((size_t)b * L_ + klr) * D_ + d0 + kdc;   // + l0 rows at use

  v8f acc[4] = {};
  for (int l0 = 0; l0 < L_; l0 += 32) {
    __syncthreads();                                    // prior tile fully consumed / sDen visible
    if (t < 32) {
      sm[t]  = colm[(size_t)b * L_ + l0 + t];
      srz[t] = 1.0f / colz[(size_t)b * L_ + l0 + t];
    }
    __syncthreads();                                    // stats visible
    // issue key-tile global loads (coalesced along d), 4 chunks per thread
    union { uint2 u; unsigned short h[4]; } kc[4];
#pragma unroll
    for (int i = 0; i < 4; ++i)
      kc[i].u = *(const uint2*)(gK + (size_t)(l0 + 8 * i) * D_);
    // attn tile: 8 contiguous floats per thread (2x float4), one shared atomic per thread
    {
      const float4 v0 = *(const float4*)(dRow + l0);
      const float4 v1 = *(const float4*)(dRow + l0 + 4);
      float vals[8] = {v0.x, v0.y, v0.z, v0.w, v1.x, v1.y, v1.z, v1.w};
      float partial = 0.f;
      _Float16* arow_lds = sAttn + arow * PAD + ac0;
#pragma unroll
      for (int i = 0; i < 8; ++i) {
        const int col = ac0 + i;
        float a = __expf(vals[i] * INV_T - sm[col]) * srz[col] + EPS_;
        partial += a;
        arow_lds[i] = (_Float16)a;
      }
      atomicAdd(&sDen[arow], partial);                  // ds_add_f32
    }
    // scatter key chunks transposed into LDS [d][l]
#pragma unroll
    for (int i = 0; i < 4; ++i) {
      const int lr = klr + 8 * i;
#pragma unroll
      for (int j = 0; j < 4; ++j)
        *((unsigned short*)sKeyT + (kdc + j) * PAD + lr) = kc[i].h[j];
    }
    __syncthreads();                                    // tiles ready

    v16h af, bf[4];
    {
      unsigned* au = (unsigned*)&af;
      const _Float16* base = sAttn + (kq * 16 + m) * PAD;
#pragma unroll
      for (int r = 0; r < 8; ++r) {
        const int kk = (r < 4) ? (g * 8 + r * 2) : (16 + g * 8 + (r - 4) * 2);
        au[r] = *(const unsigned*)(base + kk);
      }
    }
#pragma unroll
    for (int j = 0; j < 4; ++j) {
      unsigned* bu = (unsigned*)&bf[j];
      const _Float16* base = sKeyT + (dq * 64 + j * 16 + m) * PAD;
#pragma unroll
      for (int r = 0; r < 8; ++r) {
        const int kk = (r < 4) ? (g * 8 + r * 2) : (16 + g * 8 + (r - 4) * 2);
        bu[r] = *(const unsigned*)(base + kk);
      }
    }
#pragma unroll
    for (int j = 0; j < 4; ++j)
      acc[j] = __builtin_amdgcn_wmma_f32_16x16x32_f16(
          false, af, false, bf[j], (short)0, acc[j], false, false);
  }
  __syncthreads();                                      // sDen complete
  // epilogue: scale by 1/denom and store
#pragma unroll
  for (int r = 0; r < 8; ++r) {
    const int krel = kq * 16 + g * 8 + r;
    const float scale = 1.0f / sDen[krel];
    const int k = k0 + krel;
#pragma unroll
    for (int j = 0; j < 4; ++j) {
      const int d = d0 + dq * 64 + j * 16 + m;
      slots[((size_t)b * K_ + k) * D_ + d] = acc[j][r] * scale;
    }
  }
}

extern "C" void kernel_launch(void* const* d_in, const int* in_sizes, int n_in,
                              void* d_out, int out_size, void* d_ws, size_t ws_size,
                              hipStream_t stream) {
  const float* key   = (const float*)d_in[0];   // [B,L,D]
  const float* proto = (const float*)d_in[1];   // [K,D]

  // workspace layout
  _Float16* Pn  = (_Float16*)d_ws;                       // K_*D_
  _Float16* Xn  = Pn  + (size_t)K_ * D_;                 // B_*L_*D_
  _Float16* Kh  = Xn  + (size_t)B_ * L_ * D_;            // B_*L_*D_
  float*    colm  = (float*)(Kh + (size_t)B_ * L_ * D_); // B_*L_
  float*    colz  = colm + (size_t)B_ * L_;

  // output layout: (slots, dots) flat
  float* slots = (float*)d_out;
  float* dots  = slots + (size_t)B_ * K_ * D_;

  proto_norm_kernel<<<K_ / 8, 256, 0, stream>>>(proto, Pn);
  key_norm_kernel<<<(B_ * L_) / 8, 256, 0, stream>>>(key, Xn, Kh);

  dots_gemm_kernel<<<dim3(L_ / 128, K_ / 64, B_), 256, 0, stream>>>(Pn, Xn, dots);

  col_softmax_kernel<<<(B_ * L_) / 256, 256, 0, stream>>>(dots, colm, colz);

  slots_gemm_kernel<<<dim3(D_ / 128, K_ / 64, B_), 256, 0, stream>>>(
      dots, Kh, colm, colz, slots);
}
